// UnifiedGraphTransformer_18081812316374
// MI455X (gfx1250) — matrix operand order repaired and checked by reading, not compiled, not run on hardware
//
#include <hip/hip_runtime.h>
#include <hip/hip_bf16.h>

// ---------------------------------------------------------------------------
// Graph Transformer (TransformerConv x3) for MI455X / gfx1250, wave32.
// Dense GEMMs: bf16 V_WMMA_F32_16X16X32_BF16, register-blocked A (1 load per
// wave, NTILES N-tiles swept), weight panel staged to LDS via
// TENSOR_LOAD_TO_LDS (Tensor Data Mover) + s_wait_tensorcnt, B fragments read
// back with ds_load_b128. Edge softmax/aggregation is L2-resident atomic
// scatter (QKVS rows fit the 192MB L2).
// ---------------------------------------------------------------------------

typedef __bf16 v16bf __attribute__((ext_vector_type(16)));
typedef float  v8f   __attribute__((ext_vector_type(8)));
typedef unsigned int u32x4 __attribute__((ext_vector_type(4)));
typedef int          i32x4 __attribute__((ext_vector_type(4)));
typedef int          i32x8 __attribute__((ext_vector_type(8)));

#define THREADS 256
static inline int cdiv(int a, int b) { return (a + b - 1) / b; }

// ---------------------------------------------------------------- conversions
__global__ void cvt_f32_bf16_kernel(const float* __restrict__ in,
                                    __bf16* __restrict__ out, int n) {
  int t = blockIdx.x * blockDim.x + threadIdx.x;
  if (t < n) out[t] = (__bf16)in[t];
}

// W [K, Ncol] f32 row-major  ->  BT [Ncol, K] bf16 (transposed for WMMA B frag)
__global__ void pack_w_bt_kernel(const float* __restrict__ W,
                                 __bf16* __restrict__ BT, int K, int Ncol) {
  int t = blockIdx.x * blockDim.x + threadIdx.x;
  if (t >= K * Ncol) return;
  int n = t / K, k = t - n * K;
  BT[t] = (__bf16)W[k * Ncol + n];
}

// Pack Wq|Wk|Wv|Ws ([cin,128] each) into BT [512, cin] bf16 (transposed).
__global__ void pack_qkvs_bt_kernel(const float* __restrict__ Wq,
                                    const float* __restrict__ Wk,
                                    const float* __restrict__ Wv,
                                    const float* __restrict__ Ws,
                                    __bf16* __restrict__ BT, int cin) {
  int t = blockIdx.x * blockDim.x + threadIdx.x;
  if (t >= 512 * cin) return;
  int n = t / cin, k = t - n * cin;
  const float* W = (n < 128) ? Wq : (n < 256) ? Wk : (n < 384) ? Wv : Ws;
  BT[t] = (__bf16)W[k * 128 + (n & 127)];
}

__global__ void pack_bias4_kernel(const float* __restrict__ b0,
                                  const float* __restrict__ b1,
                                  const float* __restrict__ b2,
                                  const float* __restrict__ b3,
                                  float* __restrict__ out) {
  int t = blockIdx.x * blockDim.x + threadIdx.x;
  if (t >= 512) return;
  const float* b = (t < 128) ? b0 : (t < 256) ? b1 : (t < 384) ? b2 : b3;
  out[t] = b[t & 127];
}

// ---------------------------------------------------------------- WMMA GEMM
// C[M,Ncol](f32) = A[M,KT](bf16,row-major) x BT^T + bias, optional relu.
// Grid: blockIdx = (tm_block, gn). All 8 waves of a block share one column
// group of NTILES 16-wide tiles; the BT panel for that group is a contiguous
// NTILES*16*KT bf16 slab staged into LDS by the Tensor Data Mover.
// Each wave keeps its full 16xKT A strip in registers and sweeps NTILES tiles.
// Requires M % 16 == 0 (true here: N = 50000 = 3125*16) so the store epilogue
// is branch-free (no per-row exec masking).
template <int KT, int NTILES, bool RELU>
__global__ void gemm_bf16_wmma_kernel(const __bf16* __restrict__ A,
                                      const __bf16* __restrict__ BT,
                                      const float* __restrict__ bias,
                                      float* __restrict__ C, int ldc,
                                      int M, int Ncol) {
  constexpr int PANEL = NTILES * 16 * KT;  // bf16 elements, contiguous in BT
  __shared__ __bf16 Bs[PANEL];

  const int ngroups = Ncol / (16 * NTILES);
  const int gn      = blockIdx.x % ngroups;
  const int tmbase  = (blockIdx.x / ngroups) * (blockDim.x >> 5);
  const int wv      = threadIdx.x >> 5;
  const int lane    = threadIdx.x & 31;

  // ---- TDM: wave 0 DMAs the whole B panel (PANEL*2 bytes) into LDS --------
  if (wv == 0) {
    const unsigned long long ga = (unsigned long long)(BT + (size_t)gn * PANEL);
    const unsigned lds = (unsigned)(size_t)(&Bs[0]);  // flat addr[31:0] == LDS offset
    const unsigned n8  = (unsigned)(PANEL * 2 / 8);   // transfer in 8-byte elements
    // D# group0: count=1 | lds_addr | global_addr[56:0] | type=2 ("image")
    u32x4 g0 = { 1u,
                 lds,
                 (unsigned)(ga & 0xFFFFFFFFull),
                 (unsigned)((ga >> 32) & 0x1FFFFFFu) | (2u << 30) };
    // D# group1: data_size=3 (8B); 1-row tensor: dim0=tile0=stride0=n8, dim1=tile1=1
    i32x8 g1 = { (int)(3u << 16),                  // w0: data_size
                 (int)((n8 & 0xFFFFu) << 16),      // w1: tensor_dim0[15:0]
                 (int)((n8 >> 16) | (1u << 16)),   // w2: dim0[31:16] | dim1[15:0]=1
                 (int)((n8 & 0xFFFFu) << 16),      // w3: dim1[31:16]=0 | tile_dim0
                 1,                                // w4: tile_dim1=1 | tile_dim2=0
                 (int)n8,                          // w5: dim0_stride[31:0]
                 (int)((n8 & 0xFFFFu) << 16),      // w6: s0[47:32]=0 | s1[15:0]
                 (int)(n8 >> 16) };                // w7: s1[47:16]
    i32x4 z4 = {0, 0, 0, 0};
#if __clang_major__ >= 23
    i32x8 z8 = {0, 0, 0, 0, 0, 0, 0, 0};
    __builtin_amdgcn_tensor_load_to_lds(g0, g1, z4, z4, z8, 0);
#else
    __builtin_amdgcn_tensor_load_to_lds(g0, g1, z4, z4, 0);
#endif
    __builtin_amdgcn_s_wait_tensorcnt(0);
  }
  __syncthreads();

  const int tm = tmbase + wv;
  if (tm * 16 < M) {  // wave-uniform => EXEC all-ones at every WMMA
    const int mlo  = lane & 15;
    const int half = lane >> 4;
    const __bf16* Arow = A + (size_t)(tm * 16 + mlo) * KT;

    // A strip 16xKT resident in registers (ISA A-frag: K=(v>>2)*16+half*8+(v&3)*2)
    v16bf afrag[KT / 32];
#pragma unroll
    for (int kk = 0; kk < KT / 32; ++kk) {
#pragma unroll
      for (int v = 0; v < 8; ++v) {
        int ka = kk * 32 + ((v >> 2) << 4) + (half << 3) + ((v & 3) << 1);
        afrag[kk][2 * v]     = Arow[ka];
        afrag[kk][2 * v + 1] = Arow[ka + 1];
      }
    }

#pragma unroll
    for (int t = 0; t < NTILES; ++t) {
      v8f acc = {};
      const __bf16* Brow = &Bs[(size_t)(t * 16 + mlo) * KT];
#pragma unroll
      for (int kk = 0; kk < KT / 32; ++kk) {
        v16bf b{};
#pragma unroll
        for (int v = 0; v < 8; ++v) {
          // B frag: same K order as A frag: K=(v>>2)*16+half*8+(v&3)*2
          int kb = kk * 32 + ((v >> 2) << 4) + (half << 3) + ((v & 3) << 1);
          b[2 * v]     = Brow[kb];
          b[2 * v + 1] = Brow[kb + 1];
        }
        acc = __builtin_amdgcn_wmma_f32_16x16x32_bf16(
            false, afrag[kk], false, b, (short)0, acc, false, false);
      }
      // C/D layout: lane half h holds rows (tm*16 + h*8 + r), column ncol.
      const int ncol = (gn * NTILES + t) * 16 + mlo;
      const float bv = bias[ncol];
      float* Cp = C + (size_t)(tm * 16 + (half << 3)) * ldc + ncol;
#pragma unroll
      for (int r = 0; r < 8; ++r) {
        float val = acc[r] + bv;
        if (RELU) val = fmaxf(val, 0.0f);
        Cp[(size_t)r * ldc] = val;  // branch-free epilogue (M % 16 == 0)
      }
    }
  }
}

// ---------------------------------------------------------------- edge phase
__device__ __forceinline__ void atomicMaxF(float* addr, float v) {
  // signed-int max for v>=0, unsigned min for v<0: total order matches floats
  if (v >= 0.0f) atomicMax((int*)addr, __float_as_int(v));
  else           atomicMin((unsigned int*)addr, __float_as_uint(v));
}

__global__ void init_layer_kernel(float* __restrict__ amax,
                                  float* __restrict__ denom,
                                  float* __restrict__ agg, int N) {
  int t = blockIdx.x * blockDim.x + threadIdx.x;
  if (t < N * 4) { amax[t] = __int_as_float(0xff800000u); denom[t] = 0.0f; }
  if (t < N * 128) agg[t] = 0.0f;
}

// alpha[e,h] = <q[dst], k[src]>_head / sqrt(32); segment-max into amax[dst,h]
__global__ void edge_alpha_kernel(const long long* __restrict__ ei,
                                  const float* __restrict__ QKVS,
                                  float* __restrict__ alpha,
                                  float* __restrict__ amax, int E) {
  int t = blockIdx.x * blockDim.x + threadIdx.x;
  if (t >= E * 4) return;
  int e = t >> 2, h = t & 3;
  int s = (int)ei[e], d = (int)ei[E + e];
  const float4* q = (const float4*)(QKVS + (size_t)d * 512 +       h * 32);
  const float4* k = (const float4*)(QKVS + (size_t)s * 512 + 128 + h * 32);
  float acc = 0.0f;
#pragma unroll
  for (int j = 0; j < 8; ++j) {
    float4 a = q[j], b = k[j];
    acc = fmaf(a.x, b.x, fmaf(a.y, b.y, fmaf(a.z, b.z, fmaf(a.w, b.w, acc))));
  }
  acc *= 0.17677669529663687f;  // 1/sqrt(32)
  alpha[t] = acc;
  atomicMaxF(&amax[d * 4 + h], acc);
}

// ex = exp(alpha - amax[dst]); segment-sum into denom[dst,h]; ex stored in place
__global__ void edge_ex_kernel(const long long* __restrict__ ei,
                               float* __restrict__ alpha,
                               const float* __restrict__ amax,
                               float* __restrict__ denom, int E) {
  int t = blockIdx.x * blockDim.x + threadIdx.x;
  if (t >= E * 4) return;
  int e = t >> 2, h = t & 3;
  int d = (int)ei[E + e];
  float ex = expf(alpha[t] - amax[d * 4 + h]);
  alpha[t] = ex;
  atomicAdd(&denom[d * 4 + h], ex);
}

// agg[dst] += (ex/denom) * v[src]; thread = (edge, 4-channel group)
__global__ void edge_scatter_kernel(const long long* __restrict__ ei,
                                    const float* __restrict__ QKVS,
                                    const float* __restrict__ exv,
                                    const float* __restrict__ denom,
                                    float* __restrict__ agg, int E) {
  int t = blockIdx.x * blockDim.x + threadIdx.x;
  if (t >= E * 32) return;
  int e = t >> 5, g = t & 31;
  int h = g >> 3;
  int s = (int)ei[e], d = (int)ei[E + e];
  float a = exv[e * 4 + h] / (denom[d * 4 + h] + 1e-16f);
  const float4 v = *(const float4*)(QKVS + (size_t)s * 512 + 256 + g * 4);
  float* o = agg + (size_t)d * 128 + g * 4;
  atomicAdd(o + 0, a * v.x);
  atomicAdd(o + 1, a * v.y);
  atomicAdd(o + 2, a * v.z);
  atomicAdd(o + 3, a * v.w);
}

// h_next = (relu?)(agg + skip)  -> bf16 for next WMMA GEMM
__global__ void combine_kernel(const float* __restrict__ agg,
                               const float* __restrict__ QKVS,
                               __bf16* __restrict__ Hbf, int relu, int total) {
  int t = blockIdx.x * blockDim.x + threadIdx.x;
  if (t >= total) return;
  int i = t >> 7, c = t & 127;
  float v = agg[t] + QKVS[(size_t)i * 512 + 384 + c];
  if (relu) v = fmaxf(v, 0.0f);
  Hbf[t] = (__bf16)v;
}

// final tiny GEMM [N,32] x [32,10] + b2 (K=32, 10 cols — VALU is the right tool)
__global__ void cls2_kernel(const float* __restrict__ M1,
                            const float* __restrict__ W2,
                            const float* __restrict__ b2,
                            float* __restrict__ out, int N) {
  int t = blockIdx.x * blockDim.x + threadIdx.x;
  if (t >= N * 10) return;
  int i = t / 10, o = t - i * 10;
  const float* row = M1 + (size_t)i * 32;
  float acc = b2[o];
#pragma unroll
  for (int k = 0; k < 32; ++k) acc = fmaf(row[k], W2[k * 10 + o], acc);
  out[t] = acc;
}

// ---------------------------------------------------------------------------
extern "C" void kernel_launch(void* const* d_in, const int* in_sizes, int n_in,
                              void* d_out, int out_size, void* d_ws, size_t ws_size,
                              hipStream_t stream) {
  (void)n_in; (void)out_size; (void)ws_size;
  const float*     x      = (const float*)d_in[0];
  const long long* ei     = (const long long*)d_in[1];  // int64 [2,E]
  const float*     proj_W = (const float*)d_in[2];
  const float*     proj_b = (const float*)d_in[3];
  const float*     cls_W1 = (const float*)d_in[28];
  const float*     cls_b1 = (const float*)d_in[29];
  const float*     cls_W2 = (const float*)d_in[30];
  const float*     cls_b2 = (const float*)d_in[31];

  const int N = in_sizes[0] / 128;
  const int E = in_sizes[1] / 2;

  // workspace carve-out (256B aligned slabs)
  char* p = (char*)d_ws;
  auto take = [&](size_t bytes) -> char* {
    char* r = p; p += (bytes + 255) & ~(size_t)255; return r;
  };
  __bf16* Xbf   = (__bf16*)take((size_t)N * 128 * 2);
  __bf16* Hbf   = (__bf16*)take((size_t)N * 128 * 2);
  float*  QKVS  = (float*) take((size_t)N * 512 * 4);  // [q|k|v|skip] per node
  float*  agg   = (float*) take((size_t)N * 128 * 4);
  float*  amax  = (float*) take((size_t)N * 4 * 4);
  float*  denom = (float*) take((size_t)N * 4 * 4);
  float*  exbuf = (float*) take((size_t)E * 4 * 4);
  float*  M1    = (float*) take((size_t)N * 32 * 4);
  __bf16* BT    = (__bf16*)take((size_t)512 * 128 * 2);
  float*  bias4 = (float*) take(512 * 4);

  const int Mtiles = cdiv(N, 16);
  const int bm     = cdiv(Mtiles, THREADS / 32);  // row-tile blocks (8 waves each)
  int n;

  // input projection: x[N,128] -> h[N,32]
  n = N * 128;
  cvt_f32_bf16_kernel<<<cdiv(n, THREADS), THREADS, 0, stream>>>(x, Xbf, n);
  n = 128 * 32;
  pack_w_bt_kernel<<<cdiv(n, THREADS), THREADS, 0, stream>>>(proj_W, BT, 128, 32);
  gemm_bf16_wmma_kernel<128, 2, false><<<bm * 1, THREADS, 0, stream>>>(
      Xbf, BT, proj_b, M1, 32, N, 32);
  n = N * 32;
  cvt_f32_bf16_kernel<<<cdiv(n, THREADS), THREADS, 0, stream>>>(M1, Hbf, n);

  int cin = 32;
  for (int l = 0; l < 3; ++l) {
    const float* Wq = (const float*)d_in[4 + l * 8 + 0];
    const float* bq = (const float*)d_in[4 + l * 8 + 1];
    const float* Wk = (const float*)d_in[4 + l * 8 + 2];
    const float* bk = (const float*)d_in[4 + l * 8 + 3];
    const float* Wv = (const float*)d_in[4 + l * 8 + 4];
    const float* bv = (const float*)d_in[4 + l * 8 + 5];
    const float* Ws = (const float*)d_in[4 + l * 8 + 6];
    const float* bs = (const float*)d_in[4 + l * 8 + 7];

    n = 512 * cin;
    pack_qkvs_bt_kernel<<<cdiv(n, THREADS), THREADS, 0, stream>>>(Wq, Wk, Wv, Ws, BT, cin);
    pack_bias4_kernel<<<2, THREADS, 0, stream>>>(bq, bk, bv, bs, bias4);

    // fused Q|K|V|Skip GEMM: [N,cin] x [cin,512]; 4 column groups of 8 tiles
    if (cin == 32) {
      gemm_bf16_wmma_kernel<32, 8, false><<<bm * 4, THREADS, 0, stream>>>(
          Hbf, BT, bias4, QKVS, 512, N, 512);
    } else {
      gemm_bf16_wmma_kernel<128, 8, false><<<bm * 4, THREADS, 0, stream>>>(
          Hbf, BT, bias4, QKVS, 512, N, 512);
    }

    n = N * 128;
    init_layer_kernel<<<cdiv(n, THREADS), THREADS, 0, stream>>>(amax, denom, agg, N);
    n = E * 4;
    edge_alpha_kernel<<<cdiv(n, THREADS), THREADS, 0, stream>>>(ei, QKVS, exbuf, amax, E);
    edge_ex_kernel<<<cdiv(n, THREADS), THREADS, 0, stream>>>(ei, exbuf, amax, denom, E);
    n = E * 32;
    edge_scatter_kernel<<<cdiv(n, THREADS), THREADS, 0, stream>>>(ei, QKVS, exbuf, denom, agg, E);
    n = N * 128;
    combine_kernel<<<cdiv(n, THREADS), THREADS, 0, stream>>>(agg, QKVS, Hbf, (l < 2) ? 1 : 0, n);

    cin = 128;
  }

  // classifier: relu([N,128]x[128,32]+b1) then [N,32]x[32,10]+b2
  n = 128 * 32;
  pack_w_bt_kernel<<<cdiv(n, THREADS), THREADS, 0, stream>>>(cls_W1, BT, 128, 32);
  gemm_bf16_wmma_kernel<128, 2, true><<<bm * 1, THREADS, 0, stream>>>(
      Hbf, BT, cls_b1, M1, 32, N, 32);
  n = N * 10;
  cls2_kernel<<<cdiv(n, THREADS), THREADS, 0, stream>>>(M1, cls_W2, cls_b2, (float*)d_out, N);
}
